// flash_attn_2808908612118
// MI455X (gfx1250) — matrix operand order repaired and checked by reading, not compiled, not run on hardware
//
#include <hip/hip_runtime.h>

typedef __attribute__((ext_vector_type(16))) __bf16 bf16x16;
typedef __attribute__((ext_vector_type(8)))  float  floatx8;

#define E_DIM   1024
#define T_DIM   2048
#define NH_DIM  16
#define HS_DIM  64
#define THREE_E 3072

__device__ __forceinline__ unsigned short f2bf(float f) {
  union { float f; unsigned int u; } c; c.f = f;
  unsigned int u = c.u + 0x7FFFu + ((c.u >> 16) & 1u);
  return (unsigned short)(u >> 16);
}

union AFrag { bf16x16 v; unsigned int w[8]; unsigned short h[16]; };

__device__ __forceinline__ floatx8 zero8() {
  floatx8 z = {0.f,0.f,0.f,0.f,0.f,0.f,0.f,0.f};
  return z;
}

// ---- gfx1250 async copy: 16B global -> LDS, tracked by ASYNCcnt ----------
__device__ __forceinline__ void async_load_b128(const void* gaddr, void* lds) {
  // shared-aperture flat address keeps the LDS byte offset in the low 32 bits
  unsigned lds_off = (unsigned)(size_t)lds;
  asm volatile("global_load_async_to_lds_b128 %0, %1, off"
               :: "v"(lds_off), "v"(gaddr) : "memory");
}
__device__ __forceinline__ void wait_async0() {
  asm volatile("s_wait_asynccnt 0x0" ::: "memory");
}

// ---------------------------------------------------------------- convert
__global__ void cvt_f32_bf16(const float* __restrict__ in,
                             unsigned short* __restrict__ out, int n4) {
  int i = blockIdx.x * blockDim.x + threadIdx.x;
  if (i >= n4) return;
  float4 f = ((const float4*)in)[i];
  ushort4 o;
  o.x = f2bf(f.x); o.y = f2bf(f.y); o.z = f2bf(f.z); o.w = f2bf(f.w);
  ((ushort4*)out)[i] = o;
}

// ------------------------------------------------------- tiled bf16 GEMM
// C[M,N] = A[M,K] * B[K,N], 128x128 tile per block, BK=32, 8 waves.
// Each wave computes 64x32 (4x2 wmma tiles). OUT_BF16: bf16 output;
// QKV_SCALE: scale columns < E_DIM by 1/sqrt(HS) (pre-scales Q).
template<bool OUT_BF16, bool QKV_SCALE>
__global__ __launch_bounds__(256)
void gemm_bf16_128x128(const unsigned short* __restrict__ A,
                       const unsigned short* __restrict__ Bm,
                       void* __restrict__ Cout,
                       int M, int N, int K) {
  constexpr int BM = 128, BN = 128, BK = 32;
  constexpr int LDA = 40, LDB = 40;      // 80B rows: 16B aligned, bank-spread
  __shared__ unsigned short As[BM * LDA];
  __shared__ unsigned short Bs[BN * LDB];

  const int tid  = threadIdx.x;
  const int wave = tid >> 5;
  const int lane = tid & 31;
  const int g    = lane >> 4;            // lane half (0/1)
  const int ln   = lane & 15;
  const int wm   = wave >> 2;            // 0..1  (M direction)
  const int wn   = wave & 3;             // 0..3  (N direction)
  const int m0   = blockIdx.y * BM;
  const int n0   = blockIdx.x * BN;

  floatx8 acc[4][2];
#pragma unroll
  for (int a = 0; a < 4; ++a)
#pragma unroll
    for (int c = 0; c < 2; ++c) acc[a][c] = zero8();

  for (int k0 = 0; k0 < K; k0 += BK) {
    __syncthreads();
    // A tile 128x32: 512 async 16B copies straight into LDS (2 per thread)
#pragma unroll
    for (int it = 0; it < 2; ++it) {
      int c = tid + it * 256;
      int row = c >> 2, part = c & 3;
      async_load_b128(A + (size_t)(m0 + row) * K + k0 + part * 8,
                      &As[row * LDA + part * 8]);
    }
    // B tile 32x128, stored transposed Bs[n][k] (reshape -> manual staging)
    {
      int n  = tid & 127;
      int kb = (tid >> 7) * 16;
#pragma unroll
      for (int kk = 0; kk < 16; ++kk)
        Bs[n * LDB + kb + kk] = Bm[(size_t)(k0 + kb + kk) * N + n0 + n];
    }
    wait_async0();
    __syncthreads();

    AFrag af[4], bf[2];
#pragma unroll
    for (int a = 0; a < 4; ++a) {
      int row = wm * 64 + a * 16 + ln;
#pragma unroll
      for (int j = 0; j < 8; ++j) {
        int kk = 2 * j + 16 * g;
        af[a].w[j] = *(const unsigned int*)(&As[row * LDA + kk]);
      }
    }
#pragma unroll
    for (int c = 0; c < 2; ++c) {
      int col = wn * 32 + c * 16 + ln;
      const unsigned int* p = (const unsigned int*)(&Bs[col * LDB + 16 * g]);
#pragma unroll
      for (int j = 0; j < 8; ++j) bf[c].w[j] = p[j];
    }
#pragma unroll
    for (int a = 0; a < 4; ++a)
#pragma unroll
      for (int c = 0; c < 2; ++c)
        acc[a][c] = __builtin_amdgcn_wmma_f32_16x16x32_bf16(
            false, af[a].v, false, bf[c].v, (short)0, acc[a][c], false, false);
  }

#pragma unroll
  for (int a = 0; a < 4; ++a)
#pragma unroll
    for (int c = 0; c < 2; ++c) {
      int col = n0 + wn * 32 + c * 16 + ln;
#pragma unroll
      for (int i = 0; i < 8; ++i) {
        int row = m0 + wm * 64 + a * 16 + i + 8 * g;
        float v = acc[a][c][i];
        if (OUT_BF16) {
          if (QKV_SCALE && col < E_DIM) v *= 0.125f;  // 1/sqrt(64)
          ((unsigned short*)Cout)[(size_t)row * N + col] = f2bf(v);
        } else {
          ((float*)Cout)[(size_t)row * N + col] = v;
        }
      }
    }
}

// --------------------------------------------------- causal flash attention
// qkv: [B*T, 3E] bf16 (Q pre-scaled by 1/sqrt(HS)); Obuf: [B*T, E] bf16.
// Block = 128 q rows (8 waves x 16), streams 64-key tiles with online softmax.
__global__ __launch_bounds__(256)
void flash_attn_kernel(const unsigned short* __restrict__ qkv,
                       unsigned short* __restrict__ Obuf) {
  constexpr int LDK = 72;   // 144B rows: 16B aligned, bank-spread
  constexpr int LDP = 68;
  __shared__ unsigned short Ks[64 * LDK];        // K tile [key][hs]
  __shared__ unsigned short Vts[64 * LDK];       // V tile transposed [hs][key]
  __shared__ unsigned short Ps[8 * 16 * LDP];    // per-wave P staging

  const int tid  = threadIdx.x;
  const int wave = tid >> 5, lane = tid & 31;
  const int g    = lane >> 4, ln = lane & 15;
  const int qblk = blockIdx.x;                   // 0..15
  const int bh   = blockIdx.y;                   // 0..63
  const int b    = bh >> 4, h = bh & 15;

  const int    q0w     = qblk * 128 + wave * 16; // this wave's first q row
  const size_t rowBase = (size_t)b * T_DIM;

  // Q fragments (A layout), two 32-wide hs chunks, loaded once from global
  AFrag qf[2];
#pragma unroll
  for (int ch = 0; ch < 2; ++ch)
#pragma unroll
    for (int j = 0; j < 8; ++j) {
      int kk = (2 * j + 16 * g) + ch * 32;
      qf[ch].w[j] = *(const unsigned int*)(
          qkv + (rowBase + q0w + ln) * THREE_E + h * HS_DIM + kk);
    }

  // all-ones B fragment: rowsum(P) = P @ ones, lands in C layout slot i = row
  AFrag ones;
#pragma unroll
  for (int j = 0; j < 8; ++j) ones.w[j] = 0x3F803F80u;  // bf16 1.0 x2

  float m_i[8], l_i[8];
#pragma unroll
  for (int i = 0; i < 8; ++i) { m_i[i] = -1e30f; l_i[i] = 0.0f; }
  floatx8 oacc[4];
#pragma unroll
  for (int t = 0; t < 4; ++t) oacc[t] = zero8();

  const int nkt   = (qblk + 1) * 2;   // 64-key tiles needed by this block
  const int qLast = q0w + 15;

  for (int kt = 0; kt < nkt; ++kt) {
    const int k0 = kt * 64;
    __syncthreads();
    // K tile 64x64 bf16: 512 async 16B copies straight into LDS
#pragma unroll
    for (int it = 0; it < 2; ++it) {
      int c = tid + it * 256;
      int key = c >> 3, part = c & 7;
      async_load_b128(
          qkv + (rowBase + k0 + key) * THREE_E + E_DIM + h * HS_DIM + part * 8,
          &Ks[key * LDK + part * 8]);
    }
    // V tile transposed: Vts[hs][key] (reshape -> manual staging)
    {
      int hs = tid & 63;
      int kb = (tid >> 6) * 16;
#pragma unroll
      for (int kk = 0; kk < 16; ++kk)
        Vts[hs * LDK + kb + kk] =
            qkv[(rowBase + k0 + kb + kk) * THREE_E + 2 * E_DIM + h * HS_DIM + hs];
    }
    wait_async0();
    __syncthreads();

    if (k0 <= qLast) {
      const bool needMask = (k0 + 63 > qLast);   // tile straddles the diagonal
      // ---- scores: four 16x16 tiles, each K=64 over two wmma
      floatx8 s[4];
#pragma unroll
      for (int t = 0; t < 4; ++t) {
        AFrag kf0, kf1;
        int keyCol = t * 16 + ln;
        const unsigned int* p0 = (const unsigned int*)(&Ks[keyCol * LDK + 16 * g]);
        const unsigned int* p1 = (const unsigned int*)(&Ks[keyCol * LDK + 32 + 16 * g]);
#pragma unroll
        for (int j = 0; j < 8; ++j) { kf0.w[j] = p0[j]; kf1.w[j] = p1[j]; }
        floatx8 st = zero8();
        st = __builtin_amdgcn_wmma_f32_16x16x32_bf16(
            false, qf[0].v, false, kf0.v, (short)0, st, false, false);
        st = __builtin_amdgcn_wmma_f32_16x16x32_bf16(
            false, qf[1].v, false, kf1.v, (short)0, st, false, false);
        if (needMask) {
          int kg = k0 + t * 16 + ln;
#pragma unroll
          for (int i = 0; i < 8; ++i) {
            int qg = q0w + i + 8 * g;
            st[i] = (kg <= qg) ? st[i] : -1e30f;
          }
        }
        s[t] = st;
      }
      // ---- row max (rows i+8g live in same lane-half: 4 xor-shuffles)
      float rmax[8];
#pragma unroll
      for (int i = 0; i < 8; ++i) {
        float v = m_i[i];
#pragma unroll
        for (int t = 0; t < 4; ++t) v = fmaxf(v, s[t][i]);
        v = fmaxf(v, __shfl_xor(v, 1, 32));
        v = fmaxf(v, __shfl_xor(v, 2, 32));
        v = fmaxf(v, __shfl_xor(v, 4, 32));
        v = fmaxf(v, __shfl_xor(v, 8, 32));
        rmax[i] = v;
      }
#pragma unroll
      for (int i = 0; i < 8; ++i) {
        float scale = __expf(m_i[i] - rmax[i]);
        l_i[i] *= scale;
        m_i[i] = rmax[i];
#pragma unroll
        for (int t = 0; t < 4; ++t) oacc[t][i] *= scale;
      }
      // ---- P = exp(s-max): stage to LDS (C layout -> A layout transpose)
#pragma unroll
      for (int t = 0; t < 4; ++t)
#pragma unroll
        for (int i = 0; i < 8; ++i) {
          float p = __expf(s[t][i] - rmax[i]);
          Ps[(wave * 16 + i + 8 * g) * LDP + t * 16 + ln] = f2bf(p);
        }
      asm volatile("s_wait_dscnt 0" ::: "memory");  // P store -> P reload
      // ---- O += P @ V ; l += P @ ones  (rowsum via matrix core, no shuffles)
      floatx8 sacc = zero8();
#pragma unroll
      for (int kc = 0; kc < 2; ++kc) {
        AFrag pf;
#pragma unroll
        for (int j = 0; j < 8; ++j) {
          int kk = (2 * j + 16 * g) + kc * 32;
          pf.w[j] = *(const unsigned int*)(&Ps[(wave * 16 + ln) * LDP + kk]);
        }
        sacc = __builtin_amdgcn_wmma_f32_16x16x32_bf16(
            false, pf.v, false, ones.v, (short)0, sacc, false, false);
#pragma unroll
        for (int ht = 0; ht < 4; ++ht) {
          AFrag vf;
          const unsigned int* p =
              (const unsigned int*)(&Vts[(ht * 16 + ln) * LDK + kc * 32 + 16 * g]);
#pragma unroll
          for (int j = 0; j < 8; ++j) vf.w[j] = p[j];
          oacc[ht] = __builtin_amdgcn_wmma_f32_16x16x32_bf16(
              false, pf.v, false, vf.v, (short)0, oacc[ht], false, false);
        }
      }
#pragma unroll
      for (int i = 0; i < 8; ++i) l_i[i] += sacc[i];
    }
  }

  // finalize and store O as bf16 [B*T, E]
#pragma unroll
  for (int i = 0; i < 8; ++i) {
    float inv = 1.0f / l_i[i];
#pragma unroll
    for (int ht = 0; ht < 4; ++ht) oacc[ht][i] *= inv;
  }
#pragma unroll
  for (int ht = 0; ht < 4; ++ht)
#pragma unroll
    for (int i = 0; i < 8; ++i) {
      int row = q0w + i + 8 * g;
      Obuf[(rowBase + row) * E_DIM + h * HS_DIM + ht * 16 + ln] =
          f2bf(oacc[ht][i]);
    }
}

// ----------------------------------------------------------------- launch
extern "C" void kernel_launch(void* const* d_in, const int* in_sizes, int n_in,
                              void* d_out, int out_size, void* d_ws, size_t ws_size,
                              hipStream_t stream) {
  const float* x  = (const float*)d_in[0];   // [4,2048,1024]
  const float* Wa = (const float*)d_in[1];   // [1024,3072]
  const float* Wp = (const float*)d_in[2];   // [1024,1024]
  float* out = (float*)d_out;                // [4,2048,1024]

  char* ws = (char*)d_ws;
  unsigned short* xb  = (unsigned short*)ws; ws += (size_t)8192 * 1024 * 2;
  unsigned short* Wab = (unsigned short*)ws; ws += (size_t)1024 * 3072 * 2;
  unsigned short* Wpb = (unsigned short*)ws; ws += (size_t)1024 * 1024 * 2;
  unsigned short* qkv = (unsigned short*)ws; ws += (size_t)8192 * 3072 * 2;
  unsigned short* ob  = (unsigned short*)ws;

  int n4;
  n4 = 8192 * 1024 / 4;
  cvt_f32_bf16<<<(n4 + 255) / 256, 256, 0, stream>>>(x, xb, n4);
  n4 = 1024 * 3072 / 4;
  cvt_f32_bf16<<<(n4 + 255) / 256, 256, 0, stream>>>(Wa, Wab, n4);
  n4 = 1024 * 1024 / 4;
  cvt_f32_bf16<<<(n4 + 255) / 256, 256, 0, stream>>>(Wp, Wpb, n4);

  // QKV projection: [8192,1024] x [1024,3072] -> bf16, Q columns pre-scaled
  gemm_bf16_128x128<true, true>
      <<<dim3(3072 / 128, 8192 / 128), 256, 0, stream>>>(
          xb, Wab, qkv, 8192, 3072, 1024);

  // causal flash attention
  flash_attn_kernel<<<dim3(T_DIM / 128, 4 * NH_DIM), 256, 0, stream>>>(qkv, ob);

  // output projection: [8192,1024] x [1024,1024] -> f32
  gemm_bf16_128x128<false, false>
      <<<dim3(1024 / 128, 8192 / 128), 256, 0, stream>>>(
          ob, Wpb, out, 8192, 1024, 1024);
}
